// HymbaBlock_61246233641068
// MI455X (gfx1250) — compile-verified, matrix operand order checked by reading, not run on hardware
//
#include <hip/hip_runtime.h>

// ---------------------------------------------------------------------------
// Hymba block for MI455X (gfx1250, wave32, WMMA).
// All GEMMs run through a 128x128-tile LDS-staged v_wmma_f32_16x16x32_bf16
// kernel (f32 accumulate, bf16 operands), with register-prefetch software
// pipelining and global_prefetch_b8 L2 warming.  GEMM epilogue optionally
// emits f32 and/or bf16 so no separate cast passes are needed on the
// activation chain.  rope/softmax/conv/scan/eltwise are small VALU kernels.
// ---------------------------------------------------------------------------

typedef __bf16 bf16_t;
typedef __attribute__((ext_vector_type(8)))  __bf16 v8bf;
typedef __attribute__((ext_vector_type(16))) __bf16 v16bf;
typedef __attribute__((ext_vector_type(8)))  float  v8f;

#define T_SEQ   2048
#define D_MODEL 2048
#define ATTN_D  1024
#define MAMBA_D 1024
#define NHEADS  8
#define NKV     2
#define DH      128
#define DI      2048
#define NSTATE  16
#define DTRANK  64
#define DFF     8192

// ---------------- generic bf16 WMMA GEMM:  C[M,N] = A[M,K] * W[N,K]^T ------
// A row-major [M,K] bf16, W row-major [N,K] bf16.  Outputs: Cf (f32) and/or
// Cb (bf16), row-major with leading dim ldc; either may be null.
// Requires M%128==0, N%128==0, K%32==0.  256 threads = 8 wave32s; each wave
// computes a 32x64 tile = 8 x (16x16) WMMA accumulators.
#define TM 128
#define TN 128
#define TK 32

__global__ __launch_bounds__(256) void k_gemm_bf16(
    const bf16_t* __restrict__ A, const bf16_t* __restrict__ W,
    float* __restrict__ Cf, bf16_t* __restrict__ Cb,
    int M, int N, int K, int ldc)
{
  __shared__ bf16_t As[TM * TK];   // 8 KB
  __shared__ bf16_t Bs[TN * TK];   // 8 KB

  const int tid  = threadIdx.x;
  const int lane = tid & 31;
  const int wave = tid >> 5;
  const int wm   = wave >> 1;            // 0..3  (M direction, 32 rows)
  const int wn   = wave & 1;             // 0..1  (N direction, 64 cols)
  const int bm   = blockIdx.x * TM;
  const int bn   = blockIdx.y * TN;

  const int r15 = lane & 15;             // row within 16x16 frag
  const int kb  = (lane >> 4) << 3;      // K base: 0 or 8 (ISA bf16 layout)

  // global load helpers: tile is 128 rows x 4 x 16B segments = 512 segs
  const int rA0 = tid >> 2,            cA0 = (tid & 3) << 3;
  const int rA1 = (tid + 256) >> 2,    cA1 = (tid & 3) << 3;

  auto ldA = [&](int kt, int row, int c) {
    return *(const uint4*)(A + (size_t)(bm + row) * K + kt + c);
  };
  auto ldB = [&](int kt, int row, int c) {
    return *(const uint4*)(W + (size_t)(bn + row) * K + kt + c);
  };

  v8f acc[2][4] = {};

  // prologue: stage K-tile 0 into registers
  uint4 ra0 = ldA(0, rA0, cA0), ra1 = ldA(0, rA1, cA1);
  uint4 rb0 = ldB(0, rA0, cA0), rb1 = ldB(0, rA1, cA1);

  for (int kt = 0; kt < K; kt += TK) {
    ((uint4*)As)[tid]       = ra0;
    ((uint4*)As)[tid + 256] = ra1;
    ((uint4*)Bs)[tid]       = rb0;
    ((uint4*)Bs)[tid + 256] = rb1;
    __syncthreads();

    // prefetch next K-tile into registers (overlaps with WMMA below)
    int kn = kt + TK;
    if (kn < K) {
      ra0 = ldA(kn, rA0, cA0); ra1 = ldA(kn, rA1, cA1);
      rb0 = ldB(kn, rA0, cA0); rb1 = ldB(kn, rA1, cA1);
    }
    // warm GL2 two tiles ahead (global_prefetch_b8)
    int kp = kt + 2 * TK;
    if (kp < K) {
      __builtin_prefetch(A + (size_t)(bm + rA0) * K + kp + cA0, 0, 0);
      __builtin_prefetch(W + (size_t)(bn + rA0) * K + kp + cA0, 0, 0);
    }

    // --- build fragments per ISA 16-bit 16x32 layout ----------------------
    v16bf af[2], bfr[4];
#pragma unroll
    for (int i = 0; i < 2; ++i) {
      const bf16_t* p = &As[(wm * 32 + i * 16 + r15) * TK + kb];
      v8bf lo = *(const v8bf*)p;
      v8bf hi = *(const v8bf*)(p + 16);
#pragma unroll
      for (int e = 0; e < 8; ++e) { af[i][e] = lo[e]; af[i][e + 8] = hi[e]; }
    }
#pragma unroll
    for (int j = 0; j < 4; ++j) {
      const bf16_t* p = &Bs[(wn * 64 + j * 16 + r15) * TK + kb];
      v8bf lo = *(const v8bf*)p;
      v8bf hi = *(const v8bf*)(p + 16);
#pragma unroll
      for (int e = 0; e < 8; ++e) { bfr[j][e] = lo[e]; bfr[j][e + 8] = hi[e]; }
    }

#pragma unroll
    for (int i = 0; i < 2; ++i)
#pragma unroll
      for (int j = 0; j < 4; ++j)
        acc[i][j] = __builtin_amdgcn_wmma_f32_16x16x32_bf16(
            false, af[i], false, bfr[j], (short)0, acc[i][j], false, false);
    __syncthreads();
  }

  // --- store (f32 C/D layout: VGPR r -> M=r, +8 for upper lane half) ------
  const int mhi = (lane >> 4) << 3;
#pragma unroll
  for (int i = 0; i < 2; ++i)
#pragma unroll
    for (int j = 0; j < 4; ++j) {
      int row0 = bm + wm * 32 + i * 16 + mhi;
      int col  = bn + wn * 64 + j * 16 + r15;
      if (Cf) {
        float* cp = Cf + (size_t)row0 * ldc + col;
#pragma unroll
        for (int r = 0; r < 8; ++r) { *cp = acc[i][j][r]; cp += ldc; }
      }
      if (Cb) {
        bf16_t* bp = Cb + (size_t)row0 * ldc + col;
#pragma unroll
        for (int r = 0; r < 8; ++r) { *bp = (bf16_t)acc[i][j][r]; bp += ldc; }
      }
    }
}

// ---------------- elementwise / helper kernels -----------------------------

__global__ void k_cast_bf16(const float* __restrict__ s, bf16_t* __restrict__ d,
                            size_t n) {
  for (size_t i = (size_t)blockIdx.x * blockDim.x + threadIdx.x; i < n;
       i += (size_t)gridDim.x * blockDim.x)
    d[i] = (bf16_t)s[i];
}

// cast [srcRows, cols] -> [dstRows, cols], zero-padding extra rows
__global__ void k_cast_pad(const float* __restrict__ s, bf16_t* __restrict__ d,
                           int srcRows, int cols, int dstRows) {
  size_t n = (size_t)dstRows * cols;
  for (size_t i = (size_t)blockIdx.x * blockDim.x + threadIdx.x; i < n;
       i += (size_t)gridDim.x * blockDim.x) {
    int r = (int)(i / cols);
    d[i] = (r < srcRows) ? (bf16_t)s[i] : (bf16_t)0.0f;
  }
}

__global__ __launch_bounds__(256) void k_rmsnorm(
    const float* __restrict__ x, const float* __restrict__ w,
    bf16_t* __restrict__ o, int D) {
  int row = blockIdx.x;
  const float* xr = x + (size_t)row * D;
  __shared__ float red[8];
  float ss = 0.f;
  for (int i = threadIdx.x; i < D; i += 256) { float v = xr[i]; ss += v * v; }
#pragma unroll
  for (int m = 16; m >= 1; m >>= 1) ss += __shfl_xor(ss, m, 32);
  if ((threadIdx.x & 31) == 0) red[threadIdx.x >> 5] = ss;
  __syncthreads();
  float tot = 0.f;
#pragma unroll
  for (int i = 0; i < 8; ++i) tot += red[i];
  float inv = rsqrtf(tot / D + 1e-6f);
  bf16_t* orow = o + (size_t)row * D;
  for (int i = threadIdx.x; i < D; i += 256)
    orow[i] = (bf16_t)(w[i] * xr[i] * inv);
}

// rope q[T,1024] k[T,256]; emit per-head bf16 q[h][T][128], k[kv][T][128],
// and V transposed vT[kv][128][T] for the P*V GEMM.
__global__ __launch_bounds__(256) void k_rope_prep(
    const float* __restrict__ q, const float* __restrict__ k,
    const float* __restrict__ v, bf16_t* __restrict__ qo,
    bf16_t* __restrict__ ko, bf16_t* __restrict__ vT) {
  int t = blockIdx.x;
  const float LN1E4 = 9.210340372f;  // ln(10000)
  for (int p = threadIdx.x; p < 512; p += 256) {       // q: 8 heads x 64 pairs
    int h = p >> 6, i = p & 63;
    float ang = t * __expf(-(float)(2 * i) * (LN1E4 / 128.f));
    float s = __sinf(ang), c = __cosf(ang);
    const float* qr = q + (size_t)t * 1024 + h * 128;
    bf16_t* qd = qo + ((size_t)h * T_SEQ + t) * 128;
    float x1 = qr[2 * i], x2 = qr[2 * i + 1];
    qd[2 * i]     = (bf16_t)(x1 * c - x2 * s);
    qd[2 * i + 1] = (bf16_t)(x1 * s + x2 * c);
  }
  for (int p = threadIdx.x; p < 128; p += 256) {       // k: 2 kv x 64 pairs
    int h = p >> 6, i = p & 63;
    float ang = t * __expf(-(float)(2 * i) * (LN1E4 / 128.f));
    float s = __sinf(ang), c = __cosf(ang);
    const float* kr = k + (size_t)t * 256 + h * 128;
    bf16_t* kd = ko + ((size_t)h * T_SEQ + t) * 128;
    float x1 = kr[2 * i], x2 = kr[2 * i + 1];
    kd[2 * i]     = (bf16_t)(x1 * c - x2 * s);
    kd[2 * i + 1] = (bf16_t)(x1 * s + x2 * c);
  }
  for (int e = threadIdx.x; e < 256; e += 256) {       // v transpose
    int h = e >> 7, d = e & 127;
    vT[((size_t)h * 128 + d) * T_SEQ + t] = (bf16_t)v[(size_t)t * 256 + e];
  }
}

__global__ __launch_bounds__(256) void k_softmax_causal(
    const float* __restrict__ S, bf16_t* __restrict__ P, int T, float scale) {
  int r = blockIdx.x;
  const float* sr = S + (size_t)r * T;
  __shared__ float red[8];
  int nvalid = r + 1;
  float mx = -1e30f;
  for (int c = threadIdx.x; c < nvalid; c += 256)
    mx = fmaxf(mx, sr[c] * scale);
#pragma unroll
  for (int m = 16; m >= 1; m >>= 1) mx = fmaxf(mx, __shfl_xor(mx, m, 32));
  if ((threadIdx.x & 31) == 0) red[threadIdx.x >> 5] = mx;
  __syncthreads();
  float gmx = -1e30f;
#pragma unroll
  for (int i = 0; i < 8; ++i) gmx = fmaxf(gmx, red[i]);
  __syncthreads();
  float sum = 0.f;
  for (int c = threadIdx.x; c < nvalid; c += 256)
    sum += __expf(sr[c] * scale - gmx);
#pragma unroll
  for (int m = 16; m >= 1; m >>= 1) sum += __shfl_xor(sum, m, 32);
  if ((threadIdx.x & 31) == 0) red[threadIdx.x >> 5] = sum;
  __syncthreads();
  float gsum = 0.f;
#pragma unroll
  for (int i = 0; i < 8; ++i) gsum += red[i];
  float inv = 1.f / gsum;
  bf16_t* pr = P + (size_t)r * T;
  for (int c = threadIdx.x; c < T; c += 256)
    pr[c] = (c < nvalid) ? (bf16_t)(__expf(sr[c] * scale - gmx) * inv)
                         : (bf16_t)0.0f;
}

// causal depthwise conv (D_CONV=4) + bias + silu; xz row stride 4096
__global__ void k_conv_silu(const float* __restrict__ xz,
                            const float* __restrict__ cw,
                            const float* __restrict__ cb,
                            float* __restrict__ xa, bf16_t* __restrict__ xab) {
  int idx = blockIdx.x * 256 + threadIdx.x;
  if (idx >= T_SEQ * DI) return;
  int t = idx >> 11, d = idx & 2047;
  float acc = cb[d];
#pragma unroll
  for (int j = 0; j < 4; ++j) {
    int tt = t - 3 + j;
    if (tt >= 0) acc += xz[(size_t)tt * 4096 + d] * cw[d * 4 + j];
  }
  float s = acc / (1.f + __expf(-acc));
  xa[idx] = s;
  xab[idx] = (bf16_t)s;
}

// extract first DTRANK cols of padded x_dbl [T,128] -> bf16 [T,64]
__global__ void k_extract_dt(const float* __restrict__ xdbl,
                             bf16_t* __restrict__ o) {
  int idx = blockIdx.x * 256 + threadIdx.x;
  if (idx >= T_SEQ * DTRANK) return;
  int t = idx >> 6, i = idx & 63;
  o[idx] = (bf16_t)xdbl[(size_t)t * 128 + i];
}

// selective scan: thread = (channel d, state n); 16 states in a 16-lane
// half-wave, reduced with shfl_xor masks 1/2/4/8 (wave32 safe).
__global__ __launch_bounds__(256) void k_scan(
    const float* __restrict__ dtlin, const float* __restrict__ bdt,
    const float* __restrict__ xdbl, const float* __restrict__ xa,
    const float* __restrict__ xz, const float* __restrict__ Alog,
    const float* __restrict__ Dm, bf16_t* __restrict__ y) {
  int gid = blockIdx.x * 256 + threadIdx.x;  // DI*16 threads
  int d = gid >> 4, n = gid & 15;
  float A = -__expf(Alog[d * 16 + n]);
  float bd = bdt[d], Dd = Dm[d];
  float hs = 0.f;
  for (int t = 0; t < T_SEQ; ++t) {
    float dtv = dtlin[(size_t)t * DI + d] + bd;
    dtv = (dtv > 20.f) ? dtv : log1pf(__expf(dtv));
    float x  = xa[(size_t)t * DI + d];
    float Bn = xdbl[(size_t)t * 128 + 64 + n];
    float Cn = xdbl[(size_t)t * 128 + 80 + n];
    hs = __expf(dtv * A) * hs + dtv * x * Bn;
    float p = hs * Cn;
#pragma unroll
    for (int m = 8; m >= 1; m >>= 1) p += __shfl_xor(p, m, 32);
    if (n == 0) {
      float z  = xz[(size_t)t * 4096 + 2048 + d];
      float yy = p + Dd * x;
      y[(size_t)t * DI + d] = (bf16_t)(yy * (z / (1.f + __expf(-z))));
    }
  }
}

__global__ void k_residual(const float* __restrict__ x,
                           const float* __restrict__ a,
                           const float* __restrict__ m,
                           const float* __restrict__ gate,
                           float* __restrict__ xr, size_t n) {
  float g = gate[0];
  for (size_t i = (size_t)blockIdx.x * blockDim.x + threadIdx.x; i < n;
       i += (size_t)gridDim.x * blockDim.x)
    xr[i] = x[i] + g * a[i] + (1.f - g) * m[i];
}

__global__ void k_swiglu(const float* __restrict__ u1,
                         const float* __restrict__ u2,
                         bf16_t* __restrict__ g, size_t n) {
  for (size_t i = (size_t)blockIdx.x * blockDim.x + threadIdx.x; i < n;
       i += (size_t)gridDim.x * blockDim.x) {
    float a = u1[i];
    g[i] = (bf16_t)((a / (1.f + __expf(-a))) * u2[i]);
  }
}

__global__ void k_add(const float* __restrict__ a, const float* __restrict__ b,
                      float* __restrict__ o, size_t n) {
  for (size_t i = (size_t)blockIdx.x * blockDim.x + threadIdx.x; i < n;
       i += (size_t)gridDim.x * blockDim.x)
    o[i] = a[i] + b[i];
}

// ---------------------------------------------------------------------------
extern "C" void kernel_launch(void* const* d_in, const int* in_sizes, int n_in,
                              void* d_out, int out_size, void* d_ws,
                              size_t ws_size, hipStream_t stream) {
  const float* x    = (const float*)d_in[0];
  const float* n1w  = (const float*)d_in[1];
  const float* n2w  = (const float*)d_in[2];
  const float* gate = (const float*)d_in[3];
  const float* Wta  = (const float*)d_in[4];
  const float* Wq   = (const float*)d_in[5];
  const float* Wk   = (const float*)d_in[6];
  const float* Wv   = (const float*)d_in[7];
  const float* Wo   = (const float*)d_in[8];
  const float* Wpa  = (const float*)d_in[9];
  const float* Wtm  = (const float*)d_in[10];
  const float* Win  = (const float*)d_in[11];
  const float* cw   = (const float*)d_in[12];
  const float* cb   = (const float*)d_in[13];
  const float* Wx   = (const float*)d_in[14];
  const float* Wdt  = (const float*)d_in[15];
  const float* bdt  = (const float*)d_in[16];
  const float* Alog = (const float*)d_in[17];
  const float* Dm   = (const float*)d_in[18];
  const float* Wout = (const float*)d_in[19];
  const float* Wpm  = (const float*)d_in[20];
  const float* w1   = (const float*)d_in[21];
  const float* w2   = (const float*)d_in[22];
  const float* w3   = (const float*)d_in[23];

  size_t off = 0;
  auto alloc = [&](size_t bytes) -> void* {
    void* p = (char*)d_ws + off;
    off += (bytes + 255) & ~(size_t)255;
    return p;
  };
  const size_t T = T_SEQ;

  // bf16 weights
  bf16_t* Wta_b  = (bf16_t*)alloc((size_t)ATTN_D * D_MODEL * 2);
  bf16_t* Wq_b   = (bf16_t*)alloc((size_t)ATTN_D * ATTN_D * 2);
  bf16_t* Wk_b   = (bf16_t*)alloc((size_t)256 * ATTN_D * 2);
  bf16_t* Wv_b   = (bf16_t*)alloc((size_t)256 * ATTN_D * 2);
  bf16_t* Wo_b   = (bf16_t*)alloc((size_t)ATTN_D * ATTN_D * 2);
  bf16_t* Wpa_b  = (bf16_t*)alloc((size_t)D_MODEL * ATTN_D * 2);
  bf16_t* Wtm_b  = (bf16_t*)alloc((size_t)MAMBA_D * D_MODEL * 2);
  bf16_t* Win_b  = (bf16_t*)alloc((size_t)2 * DI * MAMBA_D * 2);
  bf16_t* Wx_b   = (bf16_t*)alloc((size_t)128 * DI * 2);   // padded 96->128
  bf16_t* Wdt_b  = (bf16_t*)alloc((size_t)DI * DTRANK * 2);
  bf16_t* Wout_b = (bf16_t*)alloc((size_t)MAMBA_D * DI * 2);
  bf16_t* Wpm_b  = (bf16_t*)alloc((size_t)D_MODEL * MAMBA_D * 2);
  bf16_t* w1_b   = (bf16_t*)alloc((size_t)DFF * D_MODEL * 2);
  bf16_t* w2_b   = (bf16_t*)alloc((size_t)DFF * D_MODEL * 2);
  bf16_t* w3_b   = (bf16_t*)alloc((size_t)D_MODEL * DFF * 2);

  // activations
  bf16_t* hn_b   = (bf16_t*)alloc(T * D_MODEL * 2);
  bf16_t* ha_b   = (bf16_t*)alloc(T * ATTN_D * 2);
  float*  qf     = (float*) alloc(T * ATTN_D * 4);
  float*  kf     = (float*) alloc(T * 256 * 4);
  float*  vf     = (float*) alloc(T * 256 * 4);
  bf16_t* q_b    = (bf16_t*)alloc(T * ATTN_D * 2);
  bf16_t* k_b    = (bf16_t*)alloc(T * 256 * 2);
  bf16_t* vT_b   = (bf16_t*)alloc(T * 256 * 2);
  float*  Sbuf   = (float*) alloc(T * T * 4);
  bf16_t* Pbuf   = (bf16_t*)alloc(T * T * 2);
  bf16_t* acat_b = (bf16_t*)alloc(T * ATTN_D * 2);
  bf16_t* ao_b   = (bf16_t*)alloc(T * ATTN_D * 2);
  float*  a_out  = (float*) alloc(T * D_MODEL * 4);
  bf16_t* hm_b   = (bf16_t*)alloc(T * MAMBA_D * 2);
  float*  xzb    = (float*) alloc(T * 2 * DI * 4);
  float*  xa     = (float*) alloc(T * DI * 4);
  bf16_t* xa_b   = (bf16_t*)alloc(T * DI * 2);
  float*  xdbl   = (float*) alloc(T * 128 * 4);
  bf16_t* dtin_b = (bf16_t*)alloc(T * DTRANK * 2);
  float*  dtlin  = (float*) alloc(T * DI * 4);
  bf16_t* y_b    = (bf16_t*)alloc(T * DI * 2);
  bf16_t* m0_b   = (bf16_t*)alloc(T * MAMBA_D * 2);
  float*  m_out  = (float*) alloc(T * D_MODEL * 4);
  float*  xr     = (float*) alloc(T * D_MODEL * 4);
  bf16_t* hf_b   = (bf16_t*)alloc(T * D_MODEL * 2);
  float*  u1     = (float*) alloc(T * DFF * 4);
  float*  u2     = (float*) alloc(T * DFF * 4);
  bf16_t* g_b    = (bf16_t*)alloc(T * DFF * 2);
  float*  ffn    = (float*) alloc(T * D_MODEL * 4);

  auto castW = [&](const float* s, bf16_t* d, size_t n) {
    int blocks = (int)((n + 255) / 256);
    if (blocks > 65535) blocks = 65535;
    k_cast_bf16<<<blocks, 256, 0, stream>>>(s, d, n);
  };
  auto gemm = [&](const bf16_t* A, const bf16_t* W, float* Cf, bf16_t* Cb,
                  int M, int N, int K, int ldc) {
    dim3 g(M / TM, N / TN);
    k_gemm_bf16<<<g, 256, 0, stream>>>(A, W, Cf, Cb, M, N, K, ldc);
  };

  // ---- weight casts (deterministic, every call) ----
  castW(Wta, Wta_b, (size_t)ATTN_D * D_MODEL);
  castW(Wq,  Wq_b,  (size_t)ATTN_D * ATTN_D);
  castW(Wk,  Wk_b,  (size_t)256 * ATTN_D);
  castW(Wv,  Wv_b,  (size_t)256 * ATTN_D);
  castW(Wo,  Wo_b,  (size_t)ATTN_D * ATTN_D);
  castW(Wpa, Wpa_b, (size_t)D_MODEL * ATTN_D);
  castW(Wtm, Wtm_b, (size_t)MAMBA_D * D_MODEL);
  castW(Win, Win_b, (size_t)2 * DI * MAMBA_D);
  k_cast_pad<<<2048, 256, 0, stream>>>(Wx, Wx_b, 96, DI, 128);
  castW(Wdt, Wdt_b, (size_t)DI * DTRANK);
  castW(Wout, Wout_b, (size_t)MAMBA_D * DI);
  castW(Wpm,  Wpm_b,  (size_t)D_MODEL * MAMBA_D);
  castW(w1, w1_b, (size_t)DFF * D_MODEL);
  castW(w2, w2_b, (size_t)DFF * D_MODEL);
  castW(w3, w3_b, (size_t)D_MODEL * DFF);

  // ---- norm1 ----
  k_rmsnorm<<<(int)T, 256, 0, stream>>>(x, n1w, hn_b, D_MODEL);

  // ---- attention path ----
  gemm(hn_b, Wta_b, nullptr, ha_b, T, ATTN_D, D_MODEL, ATTN_D);
  gemm(ha_b, Wq_b, qf, nullptr, T, ATTN_D, ATTN_D, ATTN_D);
  gemm(ha_b, Wk_b, kf, nullptr, T, 256, ATTN_D, 256);
  gemm(ha_b, Wv_b, vf, nullptr, T, 256, ATTN_D, 256);
  k_rope_prep<<<(int)T, 256, 0, stream>>>(qf, kf, vf, q_b, k_b, vT_b);

  const float scale = 0.08838834764831845f;  // 1/sqrt(128)
  for (int h = 0; h < NHEADS; ++h) {
    int kv = h / (NHEADS / NKV);
    gemm(q_b + (size_t)h * T * DH, k_b + (size_t)kv * T * DH, Sbuf, nullptr,
         (int)T, (int)T, DH, (int)T);
    k_softmax_causal<<<(int)T, 256, 0, stream>>>(Sbuf, Pbuf, (int)T, scale);
    gemm(Pbuf, vT_b + (size_t)kv * DH * T, nullptr, acat_b + h * DH,
         (int)T, DH, (int)T, ATTN_D);
  }
  gemm(acat_b, Wo_b, nullptr, ao_b, T, ATTN_D, ATTN_D, ATTN_D);
  gemm(ao_b, Wpa_b, a_out, nullptr, T, D_MODEL, ATTN_D, D_MODEL);

  // ---- mamba path ----
  gemm(hn_b, Wtm_b, nullptr, hm_b, T, MAMBA_D, D_MODEL, MAMBA_D);
  gemm(hm_b, Win_b, xzb, nullptr, T, 2 * DI, MAMBA_D, 2 * DI);
  k_conv_silu<<<(int)((T * DI + 255) / 256), 256, 0, stream>>>(xzb, cw, cb, xa,
                                                               xa_b);
  gemm(xa_b, Wx_b, xdbl, nullptr, T, 128, DI, 128);
  k_extract_dt<<<(int)((T * DTRANK + 255) / 256), 256, 0, stream>>>(xdbl,
                                                                    dtin_b);
  gemm(dtin_b, Wdt_b, dtlin, nullptr, T, DI, DTRANK, DI);
  k_scan<<<(DI * NSTATE) / 256, 256, 0, stream>>>(dtlin, bdt, xdbl, xa, xzb,
                                                  Alog, Dm, y_b);
  gemm(y_b, Wout_b, nullptr, m0_b, T, MAMBA_D, DI, MAMBA_D);
  gemm(m0_b, Wpm_b, m_out, nullptr, T, D_MODEL, MAMBA_D, D_MODEL);

  // ---- residual + norm2 + FFN ----
  size_t nTD = T * D_MODEL;
  k_residual<<<4096, 256, 0, stream>>>(x, a_out, m_out, gate, xr, nTD);
  k_rmsnorm<<<(int)T, 256, 0, stream>>>(xr, n2w, hf_b, D_MODEL);
  gemm(hf_b, w1_b, u1, nullptr, T, DFF, D_MODEL, DFF);
  gemm(hf_b, w2_b, u2, nullptr, T, DFF, D_MODEL, DFF);
  k_swiglu<<<16384, 256, 0, stream>>>(u1, u2, g_b, T * DFF);
  gemm(g_b, w3_b, ffn, nullptr, T, D_MODEL, DFF, D_MODEL);
  k_add<<<4096, 256, 0, stream>>>(xr, ffn, (float*)d_out, nTD);

  (void)in_sizes; (void)n_in; (void)out_size; (void)ws_size;
}
